// GCNFormerBranch_19696720019999
// MI455X (gfx1250) — compile-verified, matrix operand order checked by reading, not run on hardware
//
#include <hip/hip_runtime.h>
#include <math.h>

// ---------------------------------------------------------------------------
// MotionAGFormer block for MI455X (gfx1250, wave32, WMMA).
// All heavy GEMMs (gU/gV, TCN pointwise, MLP fc1/fc2) run on
// v_wmma_f32_16x16x32_bf16 with f32 accumulation. Residual path stays fp32;
// bf16 GEMM error is attenuated by layer_scale = 1e-5 before touching d_out.
// A-operand staging uses GLOBAL_LOAD_ASYNC_TO_LDS_B128 (ASYNCcnt-tracked,
// no VGPR data movement); B staging converts fp32 weights -> bf16 in flight.
// GEMM blocks are 8 waves; each wave owns a 16x64 tile (32 acc VGPRs) so the
// inner loop (4 batched B frags + 4 WMMAs per K-step) stays spill-free.
// ---------------------------------------------------------------------------

typedef __attribute__((ext_vector_type(16))) __bf16 v16bf;
typedef __attribute__((ext_vector_type(8)))  float  v8f;

#define NROWS (32 * 243 * 17)   // 132192
#define EPSV 1e-5f

__device__ __forceinline__ unsigned short f2bf(float f) {
  unsigned u = __float_as_uint(f);
  u += 0x7FFFu + ((u >> 16) & 1u);           // round to nearest even
  return (unsigned short)(u >> 16);
}
__device__ __forceinline__ float bf2f(unsigned short h) {
  return __uint_as_float(((unsigned)h) << 16);
}

// Load a 16x32 bf16 fragment (A or B^T) from an LDS tile stored row-major
// [16 rows x ld cols]. ISA layout: lane L holds row M=L%16; VGPR r holds the
// packed K pair k0,k0+1 with k0 = (r<4?0:16) + (L/16)*8 + (r%4)*2.
__device__ __forceinline__ v16bf load_frag(const unsigned short* base, int ld, int lane) {
  int m = lane & 15, half = lane >> 4;
  union { v16bf v; unsigned u[8]; } f;
  const unsigned short* rp = base + m * ld;
#pragma unroll
  for (int r = 0; r < 8; ++r) {
    int k0 = ((r < 4) ? 0 : 16) + half * 8 + (r & 3) * 2;
    f.u[r] = *(const unsigned*)&rp[k0];      // k0 even -> 4B aligned
  }
  return f.v;
}

// ---------------------------------------------------------------------------
// prep: fold constants into ws tables.
// P layout (floats): [0,128)=zs  [128,256)=zb  [256,352)=c2s  [352,448)=c2b
//                    [448,737)=adj(17x17) [737,754)=gs [754,771)=gb
// ---------------------------------------------------------------------------
__global__ void prep_kernel(
    const float* t1_pwb, const float* t1_bn1w, const float* t1_bn1b,
    const float* t2_pwb, const float* t2_bn1w, const float* t2_bn1b,
    const float* t3_pwb, const float* t3_bn1w, const float* t3_bn1b,
    const float* t4_pwb, const float* t4_bnw,  const float* t4_bnb,
    const float* t1_cb,  const float* t1_bn2w, const float* t1_bn2b,
    const float* t2_cb,  const float* t2_bn2w, const float* t2_bn2b,
    const float* t3_bn2w, const float* t3_bn2b,
    const float* gbn_w,  const float* gbn_b,
    float* P) {
  int tid = threadIdx.x;                  // 128 threads
  float rs = 1.0f / sqrtf(1.0f + EPSV);   // eval BN scale: w / sqrt(1+eps)

  {  // pointwise epilogue scale/bias (z = relu?(acc*zs + zb))
    int br = tid >> 5, cc = tid & 31;
    float pb, w, b;
    if (br == 0)      { pb = t1_pwb[cc]; w = t1_bn1w[cc]; b = t1_bn1b[cc]; }
    else if (br == 1) { pb = t2_pwb[cc]; w = t2_bn1w[cc]; b = t2_bn1b[cc]; }
    else if (br == 2) { pb = t3_pwb[cc]; w = t3_bn1w[cc]; b = t3_bn1b[cc]; }
    else              { pb = t4_pwb[cc]; w = t4_bnw[cc];  b = t4_bnb[cc]; }
    float s = w * rs;
    P[tid] = s;
    P[128 + tid] = pb * s + b;
  }
  if (tid < 96) {  // post-tconv / post-pool BN2 (conv bias folded in)
    int br = tid >> 5, cc = tid & 31;
    float s, b, cbv = 0.f;
    if (br == 0)      { s = t1_bn2w[cc] * rs; b = t1_bn2b[cc]; cbv = t1_cb[cc]; }
    else if (br == 1) { s = t2_bn2w[cc] * rs; b = t2_bn2b[cc]; cbv = t2_cb[cc]; }
    else              { s = t3_bn2w[cc] * rs; b = t3_bn2b[cc]; }
    P[256 + tid] = s;
    P[352 + tid] = cbv * s + b;
  }
  // normalized adjacency D^-1/2 A D^-1/2 from the H36M edge list
  const int EU[16] = {0,1,2,0,4,5,0,7,8,9, 8,11,12, 8,14,15};
  const int EV[16] = {1,2,3,4,5,6,7,8,9,10,11,12,13,14,15,16};
  for (int i = tid; i < 289; i += 128) {
    int r = i / 17, c = i % 17;
    int e = 0, dr = 0, dc = 0;
    for (int k = 0; k < 16; ++k) {
      if ((EU[k] == r && EV[k] == c) || (EU[k] == c && EV[k] == r)) e = 1;
      if (EU[k] == r || EV[k] == r) dr++;
      if (EU[k] == c || EV[k] == c) dc++;
    }
    P[448 + i] = e ? (1.0f / sqrtf((float)dr)) * (1.0f / sqrtf((float)dc)) : 0.0f;
  }
  if (tid < 17) {
    P[737 + tid] = gbn_w[tid] * rs;
    P[754 + tid] = gbn_b[tid];
  }
}

// ---------------------------------------------------------------------------
// LayerNorm over C=128, one wave per row, output bf16.
// ---------------------------------------------------------------------------
__global__ void ln_kernel(const float* __restrict__ X, const float* __restrict__ g,
                          const float* __restrict__ bta, unsigned short* __restrict__ out,
                          int n) {
  int wv = threadIdx.x >> 5, lane = threadIdx.x & 31;
  int row = blockIdx.x * 8 + wv;
  if (row >= n) return;
  const float* xr = X + (size_t)row * 128;
  float4 v = *(const float4*)&xr[lane * 4];
  float s = v.x + v.y + v.z + v.w;
  for (int m = 16; m > 0; m >>= 1) s += __shfl_xor(s, m, 32);
  float mu = s * (1.0f / 128.0f);
  float dx = v.x - mu, dy = v.y - mu, dz = v.z - mu, dw = v.w - mu;
  float q = dx * dx + dy * dy + dz * dz + dw * dw;
  for (int m = 16; m > 0; m >>= 1) q += __shfl_xor(q, m, 32);
  float r = rsqrtf(q * (1.0f / 128.0f) + EPSV);
  int c = lane * 4;
  float4 gg = *(const float4*)&g[c];
  float4 bb = *(const float4*)&bta[c];
  unsigned short* o = out + (size_t)row * 128 + c;
  o[0] = f2bf(dx * r * gg.x + bb.x);
  o[1] = f2bf(dy * r * gg.y + bb.y);
  o[2] = f2bf(dz * r * gg.z + bb.z);
  o[3] = f2bf(dw * r * gg.w + bb.w);
}

// ---------------------------------------------------------------------------
// Generic 128-column WMMA GEMM: out[N,128] = A(bf16)[N,K] x W^T + epilogue.
// Block = 256 threads (8 waves). Waves (wv&3) pick the 16-row strip, (wv>>2)
// picks the 64-col half: each wave = 16x64 tile, 4 v8f accumulators.
// A tile staged via GLOBAL_LOAD_ASYNC_TO_LDS_B128 (ASYNCcnt), W staged via
// VGPRs with fp32->bf16 conversion. 4 B fragments batched per K-step so one
// DS wait covers 4 back-to-back WMMAs without spilling.
// EPI: 0=UV store bf16 (+bias, ld 256, col base y*128)
//      1=PW  (z = relu?(acc*zs+zb), store bf16 ld 128)
//      2=FC1 (gelu(acc+b1), store bf16 ld 512, col base y*128)
//      3=FC2 (d_out += ls2*(acc+b2), fp32, in place)
// ---------------------------------------------------------------------------
template <int EPI>
__global__ __launch_bounds__(256, 1)
void gemm_kernel(const unsigned short* __restrict__ A, int lda, int K, int n,
                 const float* __restrict__ W0, const float* __restrict__ W1,
                 const float* __restrict__ W2, const float* __restrict__ W3,
                 const float* __restrict__ bias0, const float* __restrict__ bias1,
                 const float* __restrict__ e0, const float* __restrict__ e1,
                 unsigned short* __restrict__ outB, int ldo,
                 float* __restrict__ outF) {
  __shared__ __align__(16) unsigned short ldsA[64 * 128];    // 16 KB
  __shared__ __align__(16) unsigned short ldsB[128 * 128];   // 32 KB
  const int tid = threadIdx.x;
  const int lane = tid & 31, wv = tid >> 5;
  const int mw = (wv & 3) * 16;        // row strip within 64-row supertile
  const int ng = (wv >> 2) * 64;       // column half within 128 cols
  const int m0 = blockIdx.x * 64;
  const int cb = blockIdx.y * 128;

  v8f acc[4];
  const v8f vz = {0.f, 0.f, 0.f, 0.f, 0.f, 0.f, 0.f, 0.f};
#pragma unroll
  for (int i = 0; i < 4; ++i) acc[i] = vz;

  for (int kc = 0; kc < K; kc += 128) {
    // stage A: 64 rows x 128 bf16 (1024 x B128), async global -> LDS
#pragma unroll
    for (int i = 0; i < 4; ++i) {
      int u4 = tid + i * 256;
      int row = u4 >> 4;
      int c8 = (u4 & 15) * 8;
      int gr = m0 + row;
      if (gr >= n) gr = n - 1;
      unsigned long long ga =
          (unsigned long long)(uintptr_t)&A[(size_t)gr * lda + kc + c8];
      unsigned la = (unsigned)(uintptr_t)&ldsA[row * 128 + c8];
      asm volatile("global_load_async_to_lds_b128 %0, %1, off"
                   :: "v"(la), "v"(ga) : "memory");
    }
    // stage W rows (fp32 -> bf16): 128 rows x 128 cols (4096 float4)
#pragma unroll
    for (int i = 0; i < 16; ++i) {
      int f4 = tid + i * 256;
      int o = f4 >> 5;
      int c4 = (f4 & 31) * 4;
      const float* wrow;
      if (EPI == 0) {
        wrow = (blockIdx.y ? W1 : W0) + (size_t)o * K;
      } else if (EPI == 1) {
        const float* ws4[4] = {W0, W1, W2, W3};
        wrow = ws4[o >> 5] + (size_t)(o & 31) * K;
      } else {
        wrow = W0 + (size_t)(cb + o) * K;
      }
      float4 w = *(const float4*)&wrow[kc + c4];
      unsigned short* dst = &ldsB[o * 128 + c4];
      dst[0] = f2bf(w.x); dst[1] = f2bf(w.y);
      dst[2] = f2bf(w.z); dst[3] = f2bf(w.w);
    }
    asm volatile("s_wait_asynccnt 0x0" ::: "memory");
    __syncthreads();

#pragma unroll
    for (int kk = 0; kk < 128; kk += 32) {
      v16bf a = load_frag(&ldsA[mw * 128 + kk], 128, lane);
      v16bf bfrag[4];
#pragma unroll
      for (int nt = 0; nt < 4; ++nt)
        bfrag[nt] = load_frag(&ldsB[(ng + nt * 16) * 128 + kk], 128, lane);
#pragma unroll
      for (int nt = 0; nt < 4; ++nt)
        acc[nt] = __builtin_amdgcn_wmma_f32_16x16x32_bf16(
            false, a, false, bfrag[nt], (short)0, acc[nt], false, false);
    }
    __syncthreads();
  }

  // epilogue (C/D layout: lane -> col = nt*16 + lane%16, row = r + (lane/16)*8)
  const int colL = lane & 15, half = lane >> 4;
#pragma unroll
  for (int nt = 0; nt < 4; ++nt) {
    int col = ng + nt * 16 + colL;       // 0..127 within this launch segment
#pragma unroll
    for (int r = 0; r < 8; ++r) {
      int row = m0 + mw + half * 8 + r;
      if (row >= n) continue;
      float v = acc[nt][r];
      if (EPI == 0) {
        v += (blockIdx.y ? bias1 : bias0)[col];
        outB[(size_t)row * ldo + cb + col] = f2bf(v);
      } else if (EPI == 1) {
        v = v * e0[col] + e1[col];
        if (col < 96) v = fmaxf(v, 0.0f);
        outB[(size_t)row * 128 + col] = f2bf(v);
      } else if (EPI == 2) {
        v += bias0[cb + col];
        v = 0.5f * v * (1.0f + erff(v * 0.70710678118654752f));
        outB[(size_t)row * ldo + cb + col] = f2bf(v);
      } else {
        v += bias0[col];
        size_t idx = (size_t)row * 128 + col;
        outF[idx] = outF[idx] + e0[col] * v;
      }
    }
  }
}

// ---------------------------------------------------------------------------
// Spatial GCN mix: h = relu(h_ln + joint_bn(adj @ v + u)); one block per (b,t).
// ---------------------------------------------------------------------------
__global__ void gcn_kernel(const unsigned short* __restrict__ hln,
                           const unsigned short* __restrict__ uv,
                           const float* __restrict__ P,
                           unsigned short* __restrict__ hout) {
  __shared__ float vl[17 * 128];
  __shared__ float adj[289];
  __shared__ float gs[17], gb[17];
  int tid = threadIdx.x;       // 128 threads, one channel each
  size_t n0 = (size_t)blockIdx.x * 17;
  for (int i = tid; i < 289; i += 128) adj[i] = P[448 + i];
  if (tid < 17) { gs[tid] = P[737 + tid]; gb[tid] = P[754 + tid]; }
#pragma unroll 1
  for (int i = 0; i < 17; ++i)
    vl[i * 128 + tid] = bf2f(uv[(n0 + i) * 256 + 128 + tid]);
  __syncthreads();
#pragma unroll 1
  for (int i = 0; i < 17; ++i) {
    float a = 0.f;
    for (int j = 0; j < 17; ++j) a += adj[i * 17 + j] * vl[j * 128 + tid];
    float u = bf2f(uv[(n0 + i) * 256 + tid]);
    float s = (a + u) * gs[i] + gb[i];
    float hl = bf2f(hln[(n0 + i) * 128 + tid]);
    hout[(n0 + i) * 128 + tid] = f2bf(fmaxf(hl + s, 0.0f));
  }
}

// ---------------------------------------------------------------------------
// Temporal convs (dil 1 and 2), max-pool branch, identity branch, concat +
// h residual, layer_scale_1, + original x  ->  x1 written to d_out (fp32).
// Grid: (B*J=544, 243/27=9); block 128 (one channel per thread, 27 t-steps).
// ---------------------------------------------------------------------------
__global__ void tcn_x1_kernel(const unsigned short* __restrict__ z,
                              const unsigned short* __restrict__ h,
                              const float* __restrict__ x,
                              const float* __restrict__ t1_cw,
                              const float* __restrict__ t2_cw,
                              const float* __restrict__ P,
                              const float* __restrict__ ls1,
                              float* __restrict__ out) {
  __shared__ float zl[31 * 128];                // t0-2 .. t0+28
  __shared__ float w1[32 * 32 * 3], w2[32 * 32 * 3];
  __shared__ float cs[96], cb2[96];
  int tid = threadIdx.x;
  int bj = blockIdx.x;
  int b = bj / 17, j = bj % 17;
  int t0 = blockIdx.y * 27;
  for (int i = tid; i < 3072; i += 128) { w1[i] = t1_cw[i]; w2[i] = t2_cw[i]; }
  if (tid < 96) { cs[tid] = P[256 + tid]; cb2[tid] = P[352 + tid]; }
  int c = tid;
#pragma unroll 1
  for (int r = 0; r < 31; ++r) {
    int t = t0 - 2 + r;
    float v = 0.0f;
    if (t >= 0 && t < 243)
      v = bf2f(z[((size_t)((b * 243 + t) * 17 + j)) * 128 + c]);
    zl[r * 128 + c] = v;  // zero pad == conv zero pad; pool vals are >=0
  }
  __syncthreads();
  int br = c >> 5, cc = c & 31;
  float lsc = ls1[c];
#pragma unroll 1
  for (int tl = 0; tl < 27; ++tl) {
    size_t n = (size_t)((b * 243 + t0 + tl) * 17 + j);
    float o;
    if (br == 0) {                       // k=3, dil=1, in ch 0..31
      float a = 0.f;
      for (int k = 0; k < 3; ++k) {
        const float* zr = &zl[(tl + 1 + k) * 128];
        for (int i2 = 0; i2 < 32; ++i2)
          a += w1[cc * 96 + i2 * 3 + k] * zr[i2];
      }
      o = a * cs[c] + cb2[c];
    } else if (br == 1) {                // k=3, dil=2, in ch 32..63
      float a = 0.f;
      for (int k = 0; k < 3; ++k) {
        const float* zr = &zl[(tl + 2 + (k - 1) * 2) * 128 + 32];
        for (int i2 = 0; i2 < 32; ++i2)
          a += w2[cc * 96 + i2 * 3 + k] * zr[i2];
      }
      o = a * cs[c] + cb2[c];
    } else if (br == 2) {                // max-pool k=3
      float m = fmaxf(fmaxf(zl[(tl + 1) * 128 + c], zl[(tl + 2) * 128 + c]),
                      zl[(tl + 3) * 128 + c]);
      o = m * cs[c] + cb2[c];
    } else {                             // identity (already BN'ed)
      o = zl[(tl + 2) * 128 + c];
    }
    float tc = o + bf2f(h[n * 128 + c]);       // + xc residual
    out[n * 128 + c] = x[n * 128 + c] + lsc * tc;
  }
}

// ---------------------------------------------------------------------------
extern "C" void kernel_launch(void* const* d_in, const int* in_sizes, int n_in,
                              void* d_out, int out_size, void* d_ws, size_t ws_size,
                              hipStream_t stream) {
  (void)in_sizes; (void)n_in; (void)out_size; (void)ws_size;
  const float* x      = (const float*)d_in[0];
  const float* ln1_g  = (const float*)d_in[1];
  const float* ln1_b  = (const float*)d_in[2];
  const float* ln2_g  = (const float*)d_in[3];
  const float* ln2_b  = (const float*)d_in[4];
  const float* gU_w   = (const float*)d_in[5];
  const float* gU_b   = (const float*)d_in[6];
  const float* gV_w   = (const float*)d_in[7];
  const float* gV_b   = (const float*)d_in[8];
  const float* gbn_w  = (const float*)d_in[9];
  const float* gbn_b  = (const float*)d_in[10];
  const float* t1_pw  = (const float*)d_in[11];
  const float* t1_pwb = (const float*)d_in[12];
  const float* t1_bn1w= (const float*)d_in[13];
  const float* t1_bn1b= (const float*)d_in[14];
  const float* t1_cw  = (const float*)d_in[15];
  const float* t1_cb  = (const float*)d_in[16];
  const float* t1_bn2w= (const float*)d_in[17];
  const float* t1_bn2b= (const float*)d_in[18];
  const float* t2_pw  = (const float*)d_in[19];
  const float* t2_pwb = (const float*)d_in[20];
  const float* t2_bn1w= (const float*)d_in[21];
  const float* t2_bn1b= (const float*)d_in[22];
  const float* t2_cw  = (const float*)d_in[23];
  const float* t2_cb  = (const float*)d_in[24];
  const float* t2_bn2w= (const float*)d_in[25];
  const float* t2_bn2b= (const float*)d_in[26];
  const float* t3_pw  = (const float*)d_in[27];
  const float* t3_pwb = (const float*)d_in[28];
  const float* t3_bn1w= (const float*)d_in[29];
  const float* t3_bn1b= (const float*)d_in[30];
  const float* t3_bn2w= (const float*)d_in[31];
  const float* t3_bn2b= (const float*)d_in[32];
  const float* t4_pw  = (const float*)d_in[33];
  const float* t4_pwb = (const float*)d_in[34];
  const float* t4_bnw = (const float*)d_in[35];
  const float* t4_bnb = (const float*)d_in[36];
  const float* ls1    = (const float*)d_in[37];
  const float* ls2    = (const float*)d_in[38];
  const float* mlp_w1 = (const float*)d_in[39];
  const float* mlp_b1 = (const float*)d_in[40];
  const float* mlp_w2 = (const float*)d_in[41];
  const float* mlp_b2 = (const float*)d_in[42];

  float* out = (float*)d_out;
  const int N = NROWS;

  // workspace layout (all offsets multiples of 256)
  char* ws = (char*)d_ws;
  const size_t szA = (size_t)N * 128 * 2;   // bf16 [N,128]
  const size_t szB = (size_t)N * 256 * 2;   // bf16 [N,256]
  const size_t szG = (size_t)N * 512 * 2;   // bf16 [N,512]
  unsigned short* bufA = (unsigned short*)(ws);                         // h_ln, then z
  unsigned short* bufB = (unsigned short*)(ws + szA);                   // [u|v]
  unsigned short* bufC = (unsigned short*)(ws + szA + szB);             // h, then m_ln
  unsigned short* bufG = (unsigned short*)(ws + szA + szB + szA);       // gelu(fc1)
  float*          bufP = (float*)(ws + szA + szB + szA + szG);          // prep tables

  const int gM = (N + 63) / 64;             // 2066 row-supertiles
  const int gLN = (N + 7) / 8;

  prep_kernel<<<1, 128, 0, stream>>>(
      t1_pwb, t1_bn1w, t1_bn1b, t2_pwb, t2_bn1w, t2_bn1b,
      t3_pwb, t3_bn1w, t3_bn1b, t4_pwb, t4_bnw, t4_bnb,
      t1_cb, t1_bn2w, t1_bn2b, t2_cb, t2_bn2w, t2_bn2b,
      t3_bn2w, t3_bn2b, gbn_w, gbn_b, bufP);

  // 1) ln1 -> h_ln (bf16)
  ln_kernel<<<gLN, 256, 0, stream>>>(x, ln1_g, ln1_b, bufA, N);

  // 2) u,v = h_ln @ {gU,gV}^T  (WMMA)
  gemm_kernel<0><<<dim3(gM, 2), 256, 0, stream>>>(
      bufA, 128, 128, N, gU_w, gV_w, nullptr, nullptr,
      gU_b, gV_b, nullptr, nullptr, bufB, 256, nullptr);

  // 3) adjacency mix + joint BN + relu -> h
  gcn_kernel<<<32 * 243, 128, 0, stream>>>(bufA, bufB, bufP, bufC);

  // 4) fused 4-branch pointwise conv + BN1/relu (or BN4) -> z  (WMMA)
  gemm_kernel<1><<<dim3(gM, 1), 256, 0, stream>>>(
      bufC, 128, 128, N, t1_pw, t2_pw, t3_pw, t4_pw,
      nullptr, nullptr, bufP, bufP + 128, bufA, 128, nullptr);

  // 5) temporal convs / pool / concat / residual / ls1 -> x1 (d_out, fp32)
  tcn_x1_kernel<<<dim3(32 * 17, 9), 128, 0, stream>>>(
      bufA, bufC, x, t1_cw, t2_cw, bufP, ls1, out);

  // 6) ln2 -> m_ln (bf16)
  ln_kernel<<<gLN, 256, 0, stream>>>(out, ln2_g, ln2_b, bufC, N);

  // 7) fc1 + gelu -> G1  (WMMA, 4 column groups of 128)
  gemm_kernel<2><<<dim3(gM, 4), 256, 0, stream>>>(
      bufC, 128, 128, N, mlp_w1, nullptr, nullptr, nullptr,
      mlp_b1, nullptr, nullptr, nullptr, bufG, 512, nullptr);

  // 8) fc2, d_out = x1 + ls2*(G1 @ w2^T + b2)  (WMMA, K=512)
  gemm_kernel<3><<<dim3(gM, 1), 256, 0, stream>>>(
      bufG, 512, 512, N, mlp_w2, nullptr, nullptr, nullptr,
      mlp_b2, nullptr, ls2, nullptr, nullptr, 0, out);
}
